// NeuralNet_56934086476286
// MI455X (gfx1250) — matrix operand order repaired and checked by reading, not compiled, not run on hardware
//
#include <hip/hip_runtime.h>

typedef __attribute__((ext_vector_type(16))) __bf16 v16bf;
typedef __attribute__((ext_vector_type(8)))  float  v8f;
typedef __attribute__((ext_vector_type(8)))  int    v8i;

#define ITEMS  8      // batch items per workgroup (1 wave each)
#define XPAD   160    // attention input width padded 152 -> 160 (5 K-chunks of 32)
#define NE_PAD 240    // new_embeds width padded 234 -> 240

__device__ __forceinline__ unsigned short f2bf(float f) {
  unsigned u = __builtin_bit_cast(unsigned, f);
  u += 0x7FFFu + ((u >> 16) & 1u);            // round-to-nearest-even
  return (unsigned short)(u >> 16);
}
__device__ __forceinline__ float bf2f(unsigned short h) {
  return __builtin_bit_cast(float, ((unsigned)h) << 16);
}

__global__ __launch_bounds__(256) void grouprec_kernel(
    const int* __restrict__ group_ids, const int* __restrict__ movie_ids,
    const float* __restrict__ group_feats, const int* __restrict__ member_ids,
    const float* __restrict__ member_feats,
    const float* __restrict__ user_emb, const float* __restrict__ movie_emb,
    const float* __restrict__ group_emb,
    const float* __restrict__ aW1, const float* __restrict__ ab1,
    const float* __restrict__ aW2, const float* __restrict__ ab2,
    const float* __restrict__ pW1, const float* __restrict__ pb1,
    const float* __restrict__ pW2, const float* __restrict__ pb2,
    float* __restrict__ out)
{
  // x = [mem_e(64) | it_e(64) | member_feats(24) | zeros(8)] as bf16 per member
  __shared__ unsigned short x_lds[ITEMS][32][XPAD];   // 80 KB
  __shared__ unsigned short aW1_lds[XPAD][16];        // 5 KB, rows 152..159 zero
  __shared__ float pW1_lds[234 * 24];                 // 22 KB
  __shared__ float logits_lds[ITEMS][32];
  __shared__ float attn_lds[ITEMS][32];
  __shared__ float ne_lds[ITEMS][NE_PAD];

  const int tid  = threadIdx.x;
  const int w    = tid >> 5;          // wave id == item slot within workgroup
  const int lane = tid & 31;
  const int b    = blockIdx.x * ITEMS + w;

  // ---- cooperative staging of MLP weights into LDS ----
  for (int i = tid; i < XPAD * 16; i += 256) {
    int k = i >> 4;
    ((unsigned short*)aW1_lds)[i] = (k < 152) ? f2bf(aW1[i]) : (unsigned short)0;
  }
  for (int i = tid; i < 234 * 24; i += 256) pW1_lds[i] = pW1[i];

  // ---- per-item gather + bf16 convert: lane m owns member m ----
  {
    unsigned*    xr  = (unsigned*)&x_lds[w][lane][0];
    const int    mid = member_ids[b * 32 + lane];
    const float4* ur = (const float4*)(user_emb + (size_t)mid * 64);
#pragma unroll
    for (int i = 0; i < 16; ++i) {                     // mem_e -> cols 0..63
      float4 v = ur[i];
      xr[2 * i]     = f2bf(v.x) | ((unsigned)f2bf(v.y) << 16);
      xr[2 * i + 1] = f2bf(v.z) | ((unsigned)f2bf(v.w) << 16);
    }
    const int mv = movie_ids[b];
    const float4* mr = (const float4*)(movie_emb + (size_t)mv * 64);
#pragma unroll
    for (int i = 0; i < 16; ++i) {                     // it_e -> cols 64..127
      float4 v = mr[i];
      xr[32 + 2 * i]     = f2bf(v.x) | ((unsigned)f2bf(v.y) << 16);
      xr[32 + 2 * i + 1] = f2bf(v.z) | ((unsigned)f2bf(v.w) << 16);
    }
    const float4* fr = (const float4*)(member_feats + ((size_t)b * 32 + lane) * 24);
#pragma unroll
    for (int i = 0; i < 6; ++i) {                      // member_feats -> cols 128..151
      float4 v = fr[i];
      xr[64 + 2 * i]     = f2bf(v.x) | ((unsigned)f2bf(v.y) << 16);
      xr[64 + 2 * i + 1] = f2bf(v.z) | ((unsigned)f2bf(v.w) << 16);
    }
    xr[76] = 0; xr[77] = 0; xr[78] = 0; xr[79] = 0;    // zero pad cols 152..159
  }
  __syncthreads();

  // ---- B fragments for aW1 (32x16 bf16 per K-chunk), contiguous K order, same as A ----
  // lanes 0-15 hold K=0..15 (pairs per VGPR), lanes 16-31 hold K=16..31; N = lane&15
  v8i bfr[5];
#pragma unroll
  for (int c = 0; c < 5; ++c) {
    v8i t;
#pragma unroll
    for (int v = 0; v < 8; ++v) {
      const int k = c * 32 + ((lane & 16) ? 16 : 0) + 2 * v;
      const int n = lane & 15;
      unsigned lo = aW1_lds[k][n];
      unsigned hi = aW1_lds[k + 1][n];
      t[v] = (int)(lo | (hi << 16));
    }
    bfr[c] = t;
  }

  const float ab1n = ab1[lane & 15];
  const float aw2n = aW2[lane & 15];
  const float ab2v = ab2[0];

  // ---- attention layer 1: [32,160]x[160,16] via 2 M-tiles x 5 K-chunks of WMMA ----
#pragma unroll
  for (int t2 = 0; t2 < 2; ++t2) {
    const int row = (lane & 15) + 16 * t2;
    v8f acc = {0.f, 0.f, 0.f, 0.f, 0.f, 0.f, 0.f, 0.f};
#pragma unroll
    for (int c = 0; c < 5; ++c) {
      // A 16x32 bf16: vgpr v -> K = (lane>=16?16:0) + v*2 (pair)
      v8i a;
#pragma unroll
      for (int v = 0; v < 8; ++v) {
        const int k = c * 32 + ((lane & 16) ? 16 : 0) + 2 * v;
        a[v] = *(const int*)&x_lds[w][row][k];
      }
      acc = __builtin_amdgcn_wmma_f32_16x16x32_bf16(
          false, __builtin_bit_cast(v16bf, a),
          false, __builtin_bit_cast(v16bf, bfr[c]),
          (short)0, acc, false, false);
    }
    // layer 2 fused: relu(h)+bias times aW2[n], reduce over N (16 lanes per half)
    float p[8];
#pragma unroll
    for (int r = 0; r < 8; ++r) p[r] = fmaxf(acc[r] + ab1n, 0.f) * aw2n;
#pragma unroll
    for (int mk = 1; mk <= 8; mk <<= 1) {
#pragma unroll
      for (int r = 0; r < 8; ++r) p[r] += __shfl_xor(p[r], mk, 32);
    }
    if ((lane & 15) == 0) {
      const int mbase = 16 * t2 + ((lane & 16) ? 8 : 0);   // C/D: lanes>=16 -> M=r+8
#pragma unroll
      for (int r = 0; r < 8; ++r) logits_lds[w][mbase + r] = p[r] + ab2v;
    }
  }

  // ---- softmax over 32 members (full-wave shuffle reduction) ----
  {
    float l = logits_lds[w][lane];
    float mx = l;
#pragma unroll
    for (int mk = 16; mk >= 1; mk >>= 1) mx = fmaxf(mx, __shfl_xor(mx, mk, 32));
    float e = __expf(l - mx);
    float s = e;
#pragma unroll
    for (int mk = 16; mk >= 1; mk >>= 1) s += __shfl_xor(s, mk, 32);
    attn_lds[w][lane] = e / s;
  }

  // ---- pooling + new_embeds assembly; lane owns feature pair (2*lane, 2*lane+1) ----
  {
    const int fp = 2 * lane;
    float g0 = 0.f, g1 = 0.f;
#pragma unroll 4
    for (int m = 0; m < 32; ++m) {
      float a = attn_lds[w][m];
      unsigned pk = *(const unsigned*)&x_lds[w][m][fp];
      g0 = fmaf(a, bf2f((unsigned short)pk), g0);
      g1 = fmaf(a, bf2f((unsigned short)(pk >> 16)), g1);
    }
    const float* ge = group_emb + (size_t)group_ids[b] * 64;
    g0 += ge[fp]; g1 += ge[fp + 1];
    unsigned ipk = *(const unsigned*)&x_lds[w][0][64 + fp];
    float i0 = bf2f((unsigned short)ipk), i1 = bf2f((unsigned short)(ipk >> 16));
    ne_lds[w][fp]       = g0 * i0;  ne_lds[w][fp + 1]   = g1 * i1;  // g*it_e
    ne_lds[w][64 + fp]  = g0;       ne_lds[w][65 + fp]  = g1;       // g
    ne_lds[w][128 + fp] = i0;       ne_lds[w][129 + fp] = i1;       // it_e
    if (lane < 12) {                                                // wf = attn@feats
      float w0 = 0.f, w1 = 0.f;
#pragma unroll 4
      for (int m = 0; m < 32; ++m) {
        float a = attn_lds[w][m];
        unsigned pk = *(const unsigned*)&x_lds[w][m][128 + fp];
        w0 = fmaf(a, bf2f((unsigned short)pk), w0);
        w1 = fmaf(a, bf2f((unsigned short)(pk >> 16)), w1);
      }
      ne_lds[w][192 + fp] = w0; ne_lds[w][193 + fp] = w1;
    } else if (lane < 30) {                                         // group_feats
      ne_lds[w][216 + (lane - 12)] = group_feats[(size_t)b * 18 + (lane - 12)];
    }
  }

  // ---- prediction MLP: lanes 0..23 each own one hidden unit ----
  {
    const int o = (lane < 24) ? lane : 0;
    float acc2 = pb1[o];
    for (int k = 0; k < 234; ++k)
      acc2 = fmaf(ne_lds[w][k], pW1_lds[k * 24 + o], acc2);
    float ph = fmaxf(acc2, 0.f) * ((lane < 24) ? pW2[lane] : 0.f);
#pragma unroll
    for (int mk = 1; mk < 32; mk <<= 1) ph += __shfl_xor(ph, mk, 32);
    if (lane == 0) out[b] = 1.f / (1.f + __expf(-(ph + pb2[0])));
  }
}

extern "C" void kernel_launch(void* const* d_in, const int* in_sizes, int n_in,
                              void* d_out, int out_size, void* d_ws, size_t ws_size,
                              hipStream_t stream) {
  (void)in_sizes; (void)n_in; (void)out_size; (void)d_ws; (void)ws_size;
  const int*   group_ids    = (const int*)d_in[0];
  const int*   movie_ids    = (const int*)d_in[1];
  const float* group_feats  = (const float*)d_in[2];
  const int*   member_ids   = (const int*)d_in[3];
  const float* member_feats = (const float*)d_in[4];
  const float* user_emb     = (const float*)d_in[5];
  const float* movie_emb    = (const float*)d_in[6];
  const float* group_emb    = (const float*)d_in[7];
  const float* aW1 = (const float*)d_in[8];
  const float* ab1 = (const float*)d_in[9];
  const float* aW2 = (const float*)d_in[10];
  const float* ab2 = (const float*)d_in[11];
  const float* pW1 = (const float*)d_in[12];
  const float* pb1 = (const float*)d_in[13];
  const float* pW2 = (const float*)d_in[14];
  const float* pb2 = (const float*)d_in[15];
  float* out = (float*)d_out;

  dim3 grid(16384 / ITEMS), block(256);
  grouprec_kernel<<<grid, block, 0, stream>>>(
      group_ids, movie_ids, group_feats, member_ids, member_feats,
      user_emb, movie_emb, group_emb,
      aW1, ab1, aW2, ab2, pW1, pb1, pW2, pb2, out);
}
